// LocalSelfAttn_63462436765732
// MI455X (gfx1250) — matrix operand order repaired and checked by reading, not compiled, not run on hardware
//
#include <hip/hip_runtime.h>

#define BB 2
#define TT 4096
#define DD 512
#define HH 8
#define DH 64
#define BAND 16

typedef __bf16 bf16;
typedef __attribute__((ext_vector_type(16))) __bf16 v16bf;
typedef __attribute__((ext_vector_type(8)))  __bf16 v8bf;
typedef __attribute__((ext_vector_type(8)))  float  v8f;
typedef __attribute__((ext_vector_type(4)))  unsigned int v4u;
typedef __attribute__((ext_vector_type(8)))  int v8i;
typedef __attribute__((ext_vector_type(4)))  int v4i;

// ---------------------------------------------------------------------------
// WMMA helper: D = A(16x32 bf16) * B(32x16 bf16) + C(16x16 f32)
// ---------------------------------------------------------------------------
__device__ __forceinline__ v8f wmma_bf(v16bf a, v16bf b, v8f c) {
  return __builtin_amdgcn_wmma_f32_16x16x32_bf16(false, a, false, b, (short)0, c,
                                                 false, false);
}

// A-fragment (16x32 bf16, row-major source, leading dim ldk):
//   lane<16 : row=m0+lane,    K = k0+{0..15}
//   lane>=16: row=m0+lane-16, K = k0+{16..31}
__device__ __forceinline__ v16bf load_a_frag(const bf16* base, int ldk, int m0,
                                             int k0, int lane) {
  int lh = lane >> 4, lr = lane & 15;
  const bf16* row = base + (size_t)(m0 + lr) * ldk + k0 + lh * 16;
  v8bf lo = *(const v8bf*)(row);
  v8bf hi = *(const v8bf*)(row + 8);
  return __builtin_shufflevector(lo, hi, 0, 1, 2, 3, 4, 5, 6, 7, 8, 9, 10, 11,
                                 12, 13, 14, 15);
}

// B-fragment (32x16 bf16). Source is W[n,k] row-major (so B[k][n] = W[n][k]):
//   lane<16 : N=n0+lane,    K = k0+{0..15}   (one contiguous 32B load)
//   lane>=16: N=n0+lane-16, K = k0+{16..31}
__device__ __forceinline__ v16bf load_b_frag(const bf16* base, int ldk, int n0,
                                             int k0, int lane) {
  int lh = lane >> 4, lr = lane & 15;
  return *(const v16bf*)(base + (size_t)(n0 + lr) * ldk + k0 + lh * 16);
}

// ---------------------------------------------------------------------------
// TDM: DMA a 2-D bf16 tile (rows x 64, row stride 64 elems) from global into
// LDS.  Descriptor built per CDNA5 ISA 8.3/8.4.  Tracked by TENSORcnt.
// ---------------------------------------------------------------------------
__device__ __forceinline__ void tdm_load_tile_2d(unsigned int lds_off,
                                                 const bf16* gsrc, int rows) {
  unsigned long long ga = (unsigned long long)(uintptr_t)gsrc;
  v4u g0;
  g0[0] = 1u;                                   // count=1, user mode
  g0[1] = lds_off;                              // lds_addr (bytes)
  g0[2] = (unsigned int)ga;                     // global_addr[31:0]
  g0[3] = (unsigned int)((ga >> 32) & 0x1ffffffu) | (2u << 30);  // addr hi | type=2
  v8i g1;
  g1[0] = 0x00010000;                           // wg_mask=0, data_size=1 (2B)
  g1[1] = (int)(64u << 16);                     // tensor_dim0 = 64
  g1[2] = (int)((unsigned int)rows << 16);      // tensor_dim1 = rows
  g1[3] = (int)(64u << 16);                     // tile_dim0 = 64
  g1[4] = rows & 0xffff;                        // tile_dim1 = rows, tile_dim2=0
  g1[5] = 64;                                   // tensor_dim0_stride = 64
  g1[6] = 0;
  g1[7] = 0;
  v4i gz = {0, 0, 0, 0};
#if __has_include(<hip/amd_detail/amd_gfx1250_TDM.h>)
  v8i gz8 = {0, 0, 0, 0, 0, 0, 0, 0};
  __builtin_amdgcn_tensor_load_to_lds(g0, g1, gz, gz, gz8, 0);  // 6-arg toolchain
#else
  __builtin_amdgcn_tensor_load_to_lds(g0, g1, gz, gz, 0);       // 5-arg toolchain
#endif
}

// ---------------------------------------------------------------------------
// Kernel 0: fp32 -> bf16 weight conversion
// ---------------------------------------------------------------------------
__global__ void cvt_kernel(const float* __restrict__ w_in,
                           const float* __restrict__ w_out,
                           bf16* __restrict__ wi, bf16* __restrict__ wo) {
  int i = blockIdx.x * 256 + threadIdx.x;
  if (i < 3 * DD * DD) wi[i] = (bf16)w_in[i];
  if (i < DD * DD) wo[i] = (bf16)w_out[i];
}

// ---------------------------------------------------------------------------
// Kernel 1: LayerNorm, fp32 in -> bf16 h
// ---------------------------------------------------------------------------
__global__ __launch_bounds__(256) void ln_kernel(const float* __restrict__ x,
                                                 const float* __restrict__ g,
                                                 const float* __restrict__ be,
                                                 bf16* __restrict__ h) {
  int row = blockIdx.x;
  const float* xr = x + (size_t)row * DD;
  int i0 = threadIdx.x, i1 = threadIdx.x + 256;
  float a = xr[i0], b = xr[i1];
  float s = a + b, ss = a * a + b * b;
#pragma unroll
  for (int m = 16; m; m >>= 1) {
    s += __shfl_xor(s, m);
    ss += __shfl_xor(ss, m);
  }
  __shared__ float rs[8], rss[8];
  int wid = threadIdx.x >> 5, lane = threadIdx.x & 31;
  if (lane == 0) { rs[wid] = s; rss[wid] = ss; }
  __syncthreads();
  s = 0.f; ss = 0.f;
#pragma unroll
  for (int i = 0; i < 8; ++i) { s += rs[i]; ss += rss[i]; }
  float mu = s * (1.0f / DD);
  float var = ss * (1.0f / DD) - mu * mu;
  float inv = rsqrtf(var + 1e-5f);
  bf16* hr = h + (size_t)row * DD;
  hr[i0] = (bf16)((a - mu) * inv * g[i0] + be[i0]);
  hr[i1] = (bf16)((b - mu) * inv * g[i1] + be[i1]);
}

// ---------------------------------------------------------------------------
// Kernel 2: QKV GEMM  C[8192,1536] = h @ w_in^T + b_in, scattered to
// head-major q/k/v [3][B][H][T][64] bf16.  One wave = 32x32 tile.
// ---------------------------------------------------------------------------
__global__ __launch_bounds__(256) void qkv_kernel(const bf16* __restrict__ h,
                                                  const bf16* __restrict__ wi,
                                                  const float* __restrict__ b_in,
                                                  bf16* __restrict__ qkv) {
  int lane = threadIdx.x & 31;
  int wid = blockIdx.x * 8 + (threadIdx.x >> 5);
  int mt = wid / 48, nt = wid % 48;
  int m0 = mt * 32, n0 = nt * 32;
  int lh = lane >> 4, lr = lane & 15;

  v8f acc[2][2] = {};
#pragma unroll 4
  for (int kt = 0; kt < 16; ++kt) {
    int k0 = kt * 32;
    __builtin_prefetch(h + (size_t)(m0 + lr) * DD + k0 + 64, 0, 3);
    v16bf a0 = load_a_frag(h, DD, m0, k0, lane);
    v16bf a1 = load_a_frag(h, DD, m0 + 16, k0, lane);
    v16bf b0 = load_b_frag(wi, DD, n0, k0, lane);
    v16bf b1 = load_b_frag(wi, DD, n0 + 16, k0, lane);
    acc[0][0] = wmma_bf(a0, b0, acc[0][0]);
    acc[0][1] = wmma_bf(a0, b1, acc[0][1]);
    acc[1][0] = wmma_bf(a1, b0, acc[1][0]);
    acc[1][1] = wmma_bf(a1, b1, acc[1][1]);
  }
#pragma unroll
  for (int mi = 0; mi < 2; ++mi)
#pragma unroll
    for (int ni = 0; ni < 2; ++ni)
#pragma unroll
      for (int e = 0; e < 8; ++e) {
        int m = m0 + mi * 16 + lh * 8 + e;
        int n = n0 + ni * 16 + lr;
        float v = acc[mi][ni][e] + b_in[n];
        int which = n >> 9;        // q/k/v
        int r = n & 511;
        int hd = r >> 6, d = r & 63;
        int bb = m >> 12, t = m & (TT - 1);
        qkv[((((size_t)which * BB + bb) * HH + hd) * TT + t) * DH + d] = (bf16)v;
      }
}

// ---------------------------------------------------------------------------
// Kernel 3: banded attention.  One wave = one (b,h,16-row) tile.
// V window tile is DMA'd into LDS by the Tensor Data Mover (overlapped with
// the QK^T WMMAs + softmax), fenced with s_wait_tensorcnt, then transposed
// LDS->LDS for the P@V B-fragments.
// ---------------------------------------------------------------------------
__global__ __launch_bounds__(128) void attn_kernel(const bf16* __restrict__ qkv,
                                                   bf16* __restrict__ o) {
  __shared__ bf16 lds_p[4][16 * 64];     //  8 KB: P tiles
  __shared__ bf16 lds_vt[4][64 * 64];    // 32 KB: V transposed
  __shared__ bf16 lds_vrow[4][48 * 64];  // 24 KB: TDM landing (row-major V)

  int warp = threadIdx.x >> 5, lane = threadIdx.x & 31;
  int lh = lane >> 4, lr = lane & 15;
  int tile = blockIdx.x * 4 + warp;   // 0..4095
  int tt = tile & 255;
  int bh = tile >> 8;                 // 0..15
  int b = bh >> 3, hd = bh & 7;
  int t0 = tt * 16;

  const size_t plane = (size_t)BB * HH * TT * DH;
  const bf16* qh = qkv + ((size_t)(b * HH + hd) * TT) * DH;
  const bf16* kh = qh + plane;
  const bf16* vh = qh + 2 * plane;

  // --- kick off async TDM load of the V window (rows = valid band rows) ---
  int w0 = t0 - BAND;                       // window start (may be <0)
  int row_lo = max(w0, 0);
  int row_hi = min(t0 + 2 * BAND, TT);      // t0+32
  int rows = row_hi - row_lo;               // 32..48
  int front = row_lo - w0;                  // leading missing rows
  bf16* VROW = lds_vrow[warp];
  unsigned int vrow_off =
      (unsigned int)(uintptr_t)(void*)VROW + (unsigned int)front * (DH * 2);
  tdm_load_tile_2d(vrow_off, vh + (size_t)row_lo * DH, rows);

  // zero-fill the clipped window rows (edge tiles only; wave-uniform branch)
  if (rows < 48) {
    for (int i = lane; i < 48 * 64; i += 32) {
      int w = i >> 6;
      if (w < front || w >= front + rows) VROW[i] = (bf16)0.f;
    }
  }

  // --- scores S = (Q * scale) @ K_window^T (overlaps the TDM DMA) ---
  v16bf aq0 = load_a_frag(qh, DH, t0, 0, lane);
  v16bf aq1 = load_a_frag(qh, DH, t0, 32, lane);
  v8f s[3] = {};
#pragma unroll
  for (int c = 0; c < 3; ++c) {
    int col = t0 - BAND + c * 16 + lr;
    int colc = min(max(col, 0), TT - 1);
    const bf16* kr = kh + (size_t)colc * DH + lh * 16;
    v16bf bk0 = *(const v16bf*)(kr);
    v16bf bk1 = *(const v16bf*)(kr + 32);
    s[c] = wmma_bf(aq0, bk0, s[c]);
    s[c] = wmma_bf(aq1, bk1, s[c]);
  }

  // --- mask + softmax (each row lives in one 16-lane half) ---
  float sval[3][8], denom[8];
#pragma unroll
  for (int e = 0; e < 8; ++e) {
    int t = t0 + lh * 8 + e;
    float mx = -1e30f;
#pragma unroll
    for (int c = 0; c < 3; ++c) {
      int col = t0 - BAND + c * 16 + lr;
      bool valid = (col >= 0) && (col < TT) && (col >= t - BAND) && (col <= t + BAND);
      float v = valid ? s[c][e] * 0.125f : -1e30f;
      sval[c][e] = v;
      mx = fmaxf(mx, v);
    }
#pragma unroll
    for (int k = 1; k < 16; k <<= 1) mx = fmaxf(mx, __shfl_xor(mx, k));
    float sum = 0.f;
#pragma unroll
    for (int c = 0; c < 3; ++c) {
      float p = __expf(sval[c][e] - mx);
      sval[c][e] = p;
      sum += p;
    }
#pragma unroll
    for (int k = 1; k < 16; k <<= 1) sum += __shfl_xor(sum, k);
    denom[e] = sum;
  }

  // --- stage P (16x64, cols 48..63 zero) into LDS ---
  bf16* P = lds_p[warp];
#pragma unroll
  for (int e = 0; e < 8; ++e) {
    int r = lh * 8 + e;
    float inv = 1.0f / denom[e];
#pragma unroll
    for (int c = 0; c < 3; ++c) P[r * 64 + c * 16 + lr] = (bf16)(sval[c][e] * inv);
    P[r * 64 + 48 + lr] = (bf16)0.f;
  }

  // --- wait for the TDM tile, then transpose VROW -> VT in LDS ---
  __builtin_amdgcn_s_wait_tensorcnt(0);
  __syncthreads();
  bf16* VT = lds_vt[warp];
  for (int i = lane; i < 64 * 64; i += 32) {
    int w = i >> 6, d = i & 63;
    bf16 v = (w < 48) ? VROW[w * 64 + d] : (bf16)0.f;
    VT[d * 64 + w] = v;
  }
  __syncthreads();

  // --- O = P @ V  (16x64) ---
  v16bf ap0 = load_a_frag(P, 64, 0, 0, lane);
  v16bf ap1 = load_a_frag(P, 64, 0, 32, lane);
  v8f oacc[4] = {};
#pragma unroll
  for (int ntile = 0; ntile < 4; ++ntile) {
    const bf16* vr = VT + (size_t)(ntile * 16 + lr) * 64 + lh * 16;
    v16bf bv0 = *(const v16bf*)(vr);
    v16bf bv1 = *(const v16bf*)(vr + 32);
    oacc[ntile] = wmma_bf(ap0, bv0, oacc[ntile]);
    oacc[ntile] = wmma_bf(ap1, bv1, oacc[ntile]);
  }

  // --- write O back to [B,T,D] layout (bf16) ---
#pragma unroll
  for (int ntile = 0; ntile < 4; ++ntile)
#pragma unroll
    for (int e = 0; e < 8; ++e) {
      int t = t0 + lh * 8 + e;
      int d = ntile * 16 + lr;
      o[((size_t)(b * TT + t)) * DD + hd * DH + d] = (bf16)oacc[ntile][e];
    }
}

// ---------------------------------------------------------------------------
// Kernel 4: output projection + bias + residual  out = x + o @ w_out^T + b_out
// ---------------------------------------------------------------------------
__global__ __launch_bounds__(256) void proj_kernel(const bf16* __restrict__ o,
                                                   const bf16* __restrict__ wo,
                                                   const float* __restrict__ b_out,
                                                   const float* __restrict__ x,
                                                   float* __restrict__ out) {
  int lane = threadIdx.x & 31;
  int wid = blockIdx.x * 8 + (threadIdx.x >> 5);
  int mt = wid >> 4, nt = wid & 15;
  int m0 = mt * 32, n0 = nt * 32;
  int lh = lane >> 4, lr = lane & 15;

  v8f acc[2][2] = {};
#pragma unroll 4
  for (int kt = 0; kt < 16; ++kt) {
    int k0 = kt * 32;
    __builtin_prefetch(o + (size_t)(m0 + lr) * DD + k0 + 64, 0, 3);
    v16bf a0 = load_a_frag(o, DD, m0, k0, lane);
    v16bf a1 = load_a_frag(o, DD, m0 + 16, k0, lane);
    v16bf b0 = load_b_frag(wo, DD, n0, k0, lane);
    v16bf b1 = load_b_frag(wo, DD, n0 + 16, k0, lane);
    acc[0][0] = wmma_bf(a0, b0, acc[0][0]);
    acc[0][1] = wmma_bf(a0, b1, acc[0][1]);
    acc[1][0] = wmma_bf(a1, b0, acc[1][0]);
    acc[1][1] = wmma_bf(a1, b1, acc[1][1]);
  }
#pragma unroll
  for (int mi = 0; mi < 2; ++mi)
#pragma unroll
    for (int ni = 0; ni < 2; ++ni)
#pragma unroll
      for (int e = 0; e < 8; ++e) {
        int m = m0 + mi * 16 + lh * 8 + e;
        int n = n0 + ni * 16 + lr;
        size_t idx = (size_t)m * DD + n;
        out[idx] = x[idx] + acc[mi][ni][e] + b_out[n];
      }
}

// ---------------------------------------------------------------------------
// Host launcher
// ---------------------------------------------------------------------------
extern "C" void kernel_launch(void* const* d_in, const int* in_sizes, int n_in,
                              void* d_out, int out_size, void* d_ws, size_t ws_size,
                              hipStream_t stream) {
  const float* x     = (const float*)d_in[0];
  const float* ln_g  = (const float*)d_in[1];
  const float* ln_b  = (const float*)d_in[2];
  const float* w_in  = (const float*)d_in[3];
  const float* b_in  = (const float*)d_in[4];
  const float* w_out = (const float*)d_in[5];
  const float* b_out = (const float*)d_in[6];
  float* out = (float*)d_out;

  char* ws = (char*)d_ws;
  // Workspace layout (bytes):
  //   h   bf16 [B*T*D]          @ 0          (8,388,608)
  //   wi  bf16 [3D*D]           @ 8,388,608  (1,572,864)
  //   wo  bf16 [D*D]            @ 9,961,472  (  524,288)
  //   qkv bf16 [3][B][H][T][64] @ 10,485,760 (25,165,824)
  //   o   bf16 [B*T*D]          @ 35,651,584 (8,388,608)   total ~42 MB
  bf16* h    = (bf16*)(ws);
  bf16* wi   = (bf16*)(ws + 8388608);
  bf16* wo   = (bf16*)(ws + 9961472);
  bf16* qkv  = (bf16*)(ws + 10485760);
  bf16* obuf = (bf16*)(ws + 35651584);

  cvt_kernel<<<3072, 256, 0, stream>>>(w_in, w_out, wi, wo);
  ln_kernel<<<BB * TT, 256, 0, stream>>>(x, ln_g, ln_b, h);
  qkv_kernel<<<(256 * 48) / 8, 256, 0, stream>>>(h, wi, b_in, qkv);
  attn_kernel<<<(BB * HH * (TT / 16)) / 4, 128, 0, stream>>>(qkv, obuf);
  proj_kernel<<<(256 * 16) / 8, 256, 0, stream>>>(obuf, wo, b_out, x, out);
}
